// EventAttention_54382875902362
// MI455X (gfx1250) — compile-verified, matrix operand order checked by reading, not run on hardware
//
#include <hip/hip_runtime.h>
#include <hip/hip_bf16.h>

// ---------------------------------------------------------------------------
// MI455X (gfx1250) implementation:
//  * All GEMMs via v_wmma_f32_16x16x32_f16 (fp16 A/B, fp32 accum, wave32).
//  * Standalone GEMM: 32(M)x128(N) block tiles, 8 waves, 2 M-tiles/wave,
//    double-buffered LDS A staging (1 barrier/K-step), K templated & unrolled,
//    global_prefetch of the next A tile.
//  * PE second linear (dominant ~21 GFLOP) fused with LN/softmax/weighted sum
//    per event: 1 workgroup (8 wave32) per event, 16x16 WMMA tile per wave.
//  * Working set (~150MB) fits the 192MB L2 -> compute bound -> WMMA everywhere.
// ---------------------------------------------------------------------------

typedef __attribute__((ext_vector_type(16))) _Float16 v16h;
typedef __attribute__((ext_vector_type(8)))  float    v8f;

union Frag16 { unsigned u[8]; v16h h; };

#define N_EVENTS 20000
#define KNN      16
#define DIM_F    256
#define A_DIM    128
#define M_DOWN   2500
#define INV_SCALE 0.08838834764831845f   // 1/sqrt(128)
#define LN_EPS    1e-5f

// VGPR->K-pair mapping for 16-bit 16x32 A/B fragments (ISA 7.12.2):
// lanes 0-15: v0..3 -> Kpairs 0..3, v4..7 -> Kpairs 8..11
// lanes16-31: v0..3 -> Kpairs 4..7, v4..7 -> Kpairs 12..15
__device__ __forceinline__ int frag_kp(int v, int hi) {
    return (v < 4 ? v : v + 4) + (hi ? 4 : 0);
}

// --------------------------- conversion kernels ----------------------------
__global__ void cvt_f16_kernel(const float* __restrict__ src,
                               _Float16* __restrict__ dst, int n) {
    int i = blockIdx.x * 256 + threadIdx.x;
    if (i < n) dst[i] = (_Float16)src[i];
}

// src [R,C] f32 row-major  ->  dst [C,R] f16 row-major (i.e. B transposed)
__global__ void cvt_t_f16_kernel(const float* __restrict__ src,
                                 _Float16* __restrict__ dst, int R, int C) {
    int i = blockIdx.x * 256 + threadIdx.x;
    if (i < R * C) {
        int r = i / C, c = i % C;
        dst[(size_t)c * R + r] = (_Float16)src[i];
    }
}

// ------------------------------ WMMA GEMM ----------------------------------
// C[M,N] = A[M,K](f16) @ B[K,N] + bias, B supplied transposed (BT[N,K]).
// Block: 256 threads (8 waves). Block tile: 32 rows x 128 cols
// (2 M-tiles per wave, double-buffered LDS A staging, K fully unrolled).
// grid.x = M/32, grid.y = N/128.
template <int KDIM, bool RELU, bool OUT_F16>
__global__ __launch_bounds__(256) void wmma_gemm_kernel(
    const _Float16* __restrict__ A, const _Float16* __restrict__ BT,
    const float* __restrict__ bias, float* __restrict__ Cf,
    _Float16* __restrict__ Ch, int Mrows, int Ncols) {
    constexpr int NT = KDIM / 32;          // K steps
    constexpr int K2 = KDIM / 2;           // u32 per A row
    __shared__ unsigned sA[2][32 * 16];    // 2 x (32 rows x 16 u32) A tiles

    const int tid  = threadIdx.x;
    const int lane = tid & 31;
    const int wave = tid >> 5;
    const int hi   = lane >> 4;
    const int l16  = lane & 15;
    const int mbase = blockIdx.x * 32;
    const int n     = blockIdx.y * 128 + wave * 16 + l16;

    const unsigned* Au = (const unsigned*)A;
    const unsigned* Bu = (const unsigned*)BT;

    // cooperative A-tile copy: 512 u32 per step, 2 per thread
    const int arow0 = tid >> 4, akp = tid & 15;     // rows 0..15
    const int arow1 = arow0 + 16;                   // rows 16..31
    const size_t abase0 = (size_t)(mbase + arow0) * K2 + akp;
    const size_t abase1 = (size_t)(mbase + arow1) * K2 + akp;

    // prologue: stage K-step 0 into buffer 0
    sA[0][arow0 * 16 + akp] = Au[abase0];
    sA[0][arow1 * 16 + akp] = Au[abase1];
    __syncthreads();

    v8f c0 = {}, c1 = {};
#pragma unroll
    for (int kt = 0; kt < NT; ++kt) {
        const unsigned* sAc = sA[kt & 1];
        if (kt + 1 < NT) {                 // stage next tile into other buffer
            sA[(kt + 1) & 1][arow0 * 16 + akp] = Au[abase0 + (kt + 1) * 16];
            sA[(kt + 1) & 1][arow1 * 16 + akp] = Au[abase1 + (kt + 1) * 16];
            if (kt + 2 < NT)               // pull tile kt+2 toward WGP cache
                __builtin_prefetch(&Au[abase0 + (kt + 2) * 16], 0, 3);
        }
        Frag16 a0, a1, b;
#pragma unroll
        for (int v = 0; v < 8; ++v) {
            int kp = frag_kp(v, hi);
            a0.u[v] = sAc[l16 * 16 + kp];
            a1.u[v] = sAc[(16 + l16) * 16 + kp];
            b.u[v]  = Bu[(size_t)n * K2 + kt * 16 + kp];
        }
        c0 = __builtin_amdgcn_wmma_f32_16x16x32_f16(false, a0.h, false, b.h,
                                                    (short)0, c0, false, false);
        c1 = __builtin_amdgcn_wmma_f32_16x16x32_f16(false, a1.h, false, b.h,
                                                    (short)0, c1, false, false);
        __syncthreads();                   // writes to next buffer visible; reads done
    }

    const float bn = bias ? bias[n] : 0.0f;
#pragma unroll
    for (int r = 0; r < 8; ++r) {
        int m0 = mbase + r + hi * 8;
        int m1 = m0 + 16;
        float v0 = c0[r] + bn, v1 = c1[r] + bn;
        if (RELU) { v0 = fmaxf(v0, 0.0f); v1 = fmaxf(v1, 0.0f); }
        if (OUT_F16) {
            Ch[(size_t)m0 * Ncols + n] = (_Float16)v0;
            Ch[(size_t)m1 * Ncols + n] = (_Float16)v1;
        } else {
            Cf[(size_t)m0 * Ncols + n] = v0;
            Cf[(size_t)m1 * Ncols + n] = v1;
        }
    }
}

// ----------------------- global-branch k/v max pool ------------------------
__global__ __launch_bounds__(128) void maxpool_kv_kernel(
    const float* __restrict__ qkv, const int* __restrict__ pair_idx,
    float* __restrict__ kmax, float* __restrict__ vmax) {
    const int m = blockIdx.x, ch = threadIdx.x;
    float km = -3.402823466e38f, vm = -3.402823466e38f;
#pragma unroll 4
    for (int k = 0; k < KNN; ++k) {
        int j = pair_idx[m * KNN + k];
        km = fmaxf(km, qkv[(size_t)j * 384 + 128 + ch]);
        vm = fmaxf(vm, qkv[(size_t)j * 384 + 256 + ch]);
    }
    kmax[m * A_DIM + ch] = km;
    vmax[m * A_DIM + ch] = vm;
}

// ------------------- fused PE-GEMM + LN + softmax + sum --------------------
// One workgroup (8 wave32) per event. Each wave owns 16 channels.
// C-tile layout: row M(=neighbor) = r + 8*(lane>=16), col N(=channel)=lane%16.
__global__ __launch_bounds__(256) void attn_fused_kernel(
    const float* __restrict__ events,       // [N,4]
    const int*   __restrict__ nbr_idx,      // [N,16] -> kv-row index
    const int*   __restrict__ ev_map,       // optional kv-row -> event row
    const float* __restrict__ qptr, int q_stride,
    const float* __restrict__ ksrc, int k_stride,
    const float* __restrict__ vsrc, int v_stride,
    const float* __restrict__ pe_w1,        // [4,128]
    const float* __restrict__ pe_b1,        // [128]
    const _Float16* __restrict__ pe_w2T,    // [128(out),128(in)] f16
    const float* __restrict__ pe_b2,        // [128]
    const float* __restrict__ fc_g, const float* __restrict__ fc_b,
    _Float16* __restrict__ cat_out, int col_off) {
    __shared__ float    s_diff[16][4];
    __shared__ int      s_nbr[16];
    __shared__ unsigned s_hid[16 * 64];     // 16 rows x 128 f16 (packed)
    __shared__ float    s_psum[8][16];      // per-wave LN partials
    __shared__ float    s_psq[8][16];

    const int i    = blockIdx.x;
    const int tid  = threadIdx.x;
    const int lane = tid & 31;
    const int wave = tid >> 5;
    const int hi   = lane >> 4;
    const int l16  = lane & 15;

    // -- step 1: neighbor indices + event-coordinate diffs ------------------
    if (tid < 16) {
        int nb = nbr_idx[i * KNN + tid];
        s_nbr[tid] = nb;
        int er = ev_map ? ev_map[nb] : nb;
#pragma unroll
        for (int c4 = 0; c4 < 4; ++c4)
            s_diff[tid][c4] = events[i * 4 + c4] - events[er * 4 + c4];
    }
    __syncthreads();

    // -- step 2: hidden = relu(diff @ w1 + b1) -> f16 in LDS ---------------
    _Float16* s_hid_h = (_Float16*)s_hid;
#pragma unroll
    for (int j = 0; j < 8; ++j) {
        int o = tid + 256 * j;              // 0..2047
        int row = o >> 7, ch = o & 127;
        float h = pe_b1[ch];
#pragma unroll
        for (int c4 = 0; c4 < 4; ++c4)
            h += s_diff[row][c4] * pe_w1[c4 * A_DIM + ch];
        s_hid_h[row * A_DIM + ch] = (_Float16)fmaxf(h, 0.0f);
    }
    __syncthreads();

    // -- step 3: pe[16,16-ch tile] = hidden @ w2 via WMMA -------------------
    const int n = wave * 16 + l16;          // this lane's channel
    const unsigned* Bu = (const unsigned*)pe_w2T;
    v8f c = {};
#pragma unroll
    for (int kt = 0; kt < 4; ++kt) {
        Frag16 a, b;
#pragma unroll
        for (int v = 0; v < 8; ++v) {
            int kp = frag_kp(v, hi);
            a.u[v] = s_hid[l16 * 64 + kt * 16 + kp];
            b.u[v] = Bu[n * 64 + kt * 16 + kp];
        }
        c = __builtin_amdgcn_wmma_f32_16x16x32_f16(false, a.h, false, b.h,
                                                   (short)0, c, false, false);
    }

    // -- step 4: t = q - kg + pe -------------------------------------------
    const float b2n = pe_b2[n];
    const float qn  = qptr[(size_t)i * q_stride + n];
    const float gn  = fc_g[n], bn = fc_b[n];
    float t[8], pe[8];
    int nb_r[8];
#pragma unroll
    for (int r = 0; r < 8; ++r) {
        int m  = r + hi * 8;
        int nb = s_nbr[m];
        nb_r[r] = nb;
        pe[r]   = c[r] + b2n;
        float kg = ksrc[(size_t)nb * k_stride + n];
        t[r] = qn - kg + pe[r];
    }

    // -- step 5: LayerNorm over 128 channels per neighbor row --------------
#pragma unroll
    for (int r = 0; r < 8; ++r) {
        float sr = t[r], sq = t[r] * t[r];
#pragma unroll
        for (int off = 1; off <= 8; off <<= 1) {
            sr += __shfl_xor(sr, off, 32);
            sq += __shfl_xor(sq, off, 32);
        }
        if (l16 == 0) {                      // deterministic partials, no atomics
            int m = r + hi * 8;
            s_psum[wave][m] = sr;
            s_psq[wave][m]  = sq;
        }
    }
    __syncthreads();

    float u[8];
#pragma unroll
    for (int r = 0; r < 8; ++r) {
        int m = r + hi * 8;
        float sr = 0.0f, sq = 0.0f;
#pragma unroll
        for (int w8 = 0; w8 < 8; ++w8) { sr += s_psum[w8][m]; sq += s_psq[w8][m]; }
        float mu  = sr * (1.0f / 128.0f);
        float var = sq * (1.0f / 128.0f) - mu * mu;
        u[r] = ((t[r] - mu) * rsqrtf(var + LN_EPS) * gn + bn) * INV_SCALE;
    }

    // -- step 6: softmax over the 16 neighbors (rows) ----------------------
    float mx = u[0];
#pragma unroll
    for (int r = 1; r < 8; ++r) mx = fmaxf(mx, u[r]);
    mx = fmaxf(mx, __shfl_xor(mx, 16, 32));  // combine rows 0-7 and 8-15
    float e[8], se = 0.0f;
#pragma unroll
    for (int r = 0; r < 8; ++r) { e[r] = __expf(u[r] - mx); se += e[r]; }
    se += __shfl_xor(se, 16, 32);
    const float inv = 1.0f / se;

    // -- step 7: out = sum_k attn * (vg + pe) ------------------------------
    float acc = 0.0f;
#pragma unroll
    for (int r = 0; r < 8; ++r) {
        float vg = vsrc[(size_t)nb_r[r] * v_stride + n];
        acc += (e[r] * inv) * (vg + pe[r]);
    }
    acc += __shfl_xor(acc, 16, 32);
    if (!hi) cat_out[(size_t)i * (2 * A_DIM) + col_off + n] = (_Float16)acc;
}

// ------------------------------- launcher ----------------------------------
extern "C" void kernel_launch(void* const* d_in, const int* in_sizes, int n_in,
                              void* d_out, int out_size, void* d_ws, size_t ws_size,
                              hipStream_t stream) {
    const float* events   = (const float*)d_in[0];
    const float* features = (const float*)d_in[1];
    const int* local_idx    = (const int*)d_in[2];
    const int* down_idx     = (const int*)d_in[3];
    const int* pair_idx     = (const int*)d_in[4];
    const int* inv_pair_idx = (const int*)d_in[5];
    const float* lqkv_w = (const float*)d_in[6];  const float* lqkv_b = (const float*)d_in[7];
    const float* lpe_w1 = (const float*)d_in[8];  const float* lpe_b1 = (const float*)d_in[9];
    const float* lpe_w2 = (const float*)d_in[10]; const float* lpe_b2 = (const float*)d_in[11];
    const float* lfc_g  = (const float*)d_in[12]; const float* lfc_b  = (const float*)d_in[13];
    const float* gqkv_w = (const float*)d_in[14]; const float* gqkv_b = (const float*)d_in[15];
    const float* gpe_w1 = (const float*)d_in[16]; const float* gpe_b1 = (const float*)d_in[17];
    const float* gpe_w2 = (const float*)d_in[18]; const float* gpe_b2 = (const float*)d_in[19];
    const float* gfc_g  = (const float*)d_in[20]; const float* gfc_b  = (const float*)d_in[21];
    const float* pw1 = (const float*)d_in[22]; const float* pb1 = (const float*)d_in[23];
    const float* pw2 = (const float*)d_in[24]; const float* pb2 = (const float*)d_in[25];
    float* out = (float*)d_out;

    // ---- workspace carve-up (~96 MB) ----
    char* wp = (char*)d_ws;
    auto carve = [&](size_t bytes) {
        void* p = (void*)wp;
        wp += (bytes + 255) & ~(size_t)255;
        return p;
    };
    _Float16* feat_h = (_Float16*)carve((size_t)N_EVENTS * DIM_F * 2);
    _Float16* lqkvT  = (_Float16*)carve((size_t)384 * 256 * 2);
    _Float16* gqkvT  = (_Float16*)carve((size_t)384 * 256 * 2);
    _Float16* lpe2T  = (_Float16*)carve((size_t)128 * 128 * 2);
    _Float16* gpe2T  = (_Float16*)carve((size_t)128 * 128 * 2);
    _Float16* pw1T   = (_Float16*)carve((size_t)256 * 256 * 2);
    _Float16* pw2T   = (_Float16*)carve((size_t)256 * 256 * 2);
    float* qkv_l = (float*)carve((size_t)N_EVENTS * 384 * 4);
    float* qkv_g = (float*)carve((size_t)N_EVENTS * 384 * 4);
    float* kmax  = (float*)carve((size_t)M_DOWN * A_DIM * 4);
    float* vmax  = (float*)carve((size_t)M_DOWN * A_DIM * 4);
    _Float16* cat_h = (_Float16*)carve((size_t)N_EVENTS * 256 * 2);
    _Float16* hid_h = (_Float16*)carve((size_t)N_EVENTS * 256 * 2);

    // ---- 1. fp16 conversions / weight transposes ----
    {
        int n = N_EVENTS * DIM_F;
        cvt_f16_kernel<<<(n + 255) / 256, 256, 0, stream>>>(features, feat_h, n);
    }
    cvt_t_f16_kernel<<<(256 * 384 + 255) / 256, 256, 0, stream>>>(lqkv_w, lqkvT, 256, 384);
    cvt_t_f16_kernel<<<(256 * 384 + 255) / 256, 256, 0, stream>>>(gqkv_w, gqkvT, 256, 384);
    cvt_t_f16_kernel<<<(128 * 128 + 255) / 256, 256, 0, stream>>>(lpe_w2, lpe2T, 128, 128);
    cvt_t_f16_kernel<<<(128 * 128 + 255) / 256, 256, 0, stream>>>(gpe_w2, gpe2T, 128, 128);
    cvt_t_f16_kernel<<<(256 * 256 + 255) / 256, 256, 0, stream>>>(pw1, pw1T, 256, 256);
    cvt_t_f16_kernel<<<(256 * 256 + 255) / 256, 256, 0, stream>>>(pw2, pw2T, 256, 256);

    // ---- 2. QKV GEMMs: [20000,256] x [256,384] ----
    {
        dim3 g(N_EVENTS / 32, 384 / 128);
        wmma_gemm_kernel<256, false, false><<<g, 256, 0, stream>>>(
            feat_h, lqkvT, lqkv_b, qkv_l, nullptr, N_EVENTS, 384);
        wmma_gemm_kernel<256, false, false><<<g, 256, 0, stream>>>(
            feat_h, gqkvT, gqkv_b, qkv_g, nullptr, N_EVENTS, 384);
    }

    // ---- 3. global-branch k/v max pool ----
    maxpool_kv_kernel<<<M_DOWN, 128, 0, stream>>>(qkv_g, pair_idx, kmax, vmax);

    // ---- 4. fused attention (local, then global) -> fp16 concat buffer ----
    attn_fused_kernel<<<N_EVENTS, 256, 0, stream>>>(
        events, local_idx, nullptr,
        qkv_l, 384, qkv_l + 128, 384, qkv_l + 256, 384,
        lpe_w1, lpe_b1, lpe2T, lpe_b2, lfc_g, lfc_b, cat_h, 0);
    attn_fused_kernel<<<N_EVENTS, 256, 0, stream>>>(
        events, inv_pair_idx, down_idx,
        qkv_g, 384, kmax, 128, vmax, 128,
        gpe_w1, gpe_b1, gpe2T, gpe_b2, gfc_g, gfc_b, cat_h, 128);

    // ---- 5. projection MLP: ReLU GEMM (fp16 out) then final GEMM (f32) ----
    {
        dim3 g(N_EVENTS / 32, 256 / 128);
        wmma_gemm_kernel<256, true, true><<<g, 256, 0, stream>>>(
            cat_h, pw1T, pb1, nullptr, hid_h, N_EVENTS, 256);
        wmma_gemm_kernel<256, false, false><<<g, 256, 0, stream>>>(
            hid_h, pw2T, pb2, out, nullptr, N_EVENTS, 256);
    }
}
